// ECHOBlock_47820165874220
// MI455X (gfx1250) — matrix operand order repaired and checked by reading, not compiled, not run on hardware
//
#include <hip/hip_runtime.h>
#include <hip/hip_bf16.h>

typedef __attribute__((ext_vector_type(16))) _Float16 v16h;
typedef __attribute__((ext_vector_type(8)))  _Float16 v8h;
typedef __attribute__((ext_vector_type(8)))  float    v8f;

#define N_NODES 10000
#define N_EDGES 128000
#define C_IN    16
#define N_DES   16
#define RM      6      // R*M
#define KCONV   96     // R*M*C
#define NBINS   25     // B
#define DB      400    // D*B
#define DBP     416    // DB padded to multiple of 32 for WMMA K-loop
#define H1D     256
#define H2D     128
#define OUTD    32
#define EPSV    1e-8f

// ---------------- workspace layout (bytes, all 256B aligned) ----------------
#define WS_AGG    0ul                    // N*96*2  f32  = 7,680,000 B
#define WS_HIST   7680000ul              // N*400*2 f32  = 32,000,000 B
#define WS_X1     39680000ul             // N*16*2  f32
#define WS_XT     40960000ul             // N*16*2  f32
#define WS_RES    42240000ul             // N*32    f32 (includes bL3)
#define WS_DESC   43520000ul             // N*416   f16 (K-padded)
#define WS_W1T    51840000ul             // 256*416 f16
#define WS_W2T    52052992ul             // 128*256 f16
#define WS_W3T    52118528ul             // 32*128  f16
#define WS_ZERO_BYTES 39680000ul         // agg + hist contiguous

// ---------------- weight prep: f32 -> f16, transposed, K-padded -------------
__global__ void prep_weights(const float* __restrict__ W1, const float* __restrict__ W2,
                             const float* __restrict__ W3,
                             _Float16* __restrict__ W1t, _Float16* __restrict__ W2t,
                             _Float16* __restrict__ W3t) {
  int i = blockIdx.x * blockDim.x + threadIdx.x;
  const int n1 = H1D * DBP, n2 = H2D * H1D, n3 = OUTD * H2D;
  if (i < n1) {
    int h = i / DBP, k = i % DBP;
    W1t[i] = (k < DB) ? (_Float16)W1[k * H1D + h] : (_Float16)0.f;
  } else if (i < n1 + n2) {
    int j = i - n1, h = j / H1D, k = j % H1D;
    W2t[j] = (_Float16)W2[k * H2D + h];
  } else if (i < n1 + n2 + n3) {
    int j = i - n1 - n2, h = j / H2D, k = j % H2D;
    W3t[j] = (_Float16)W3[k * OUTD + h];
  }
}

// ---------------- FieldConv edge scatter ------------------------------------
// thread = (edge, r*m). z = pcmp_scatter[e,rm] * (connection[e] * x[src,c])
__global__ void fieldconv_scatter(const float* __restrict__ x,
                                  const long long* __restrict__ ei,
                                  const float* __restrict__ pscat,
                                  const float* __restrict__ conn,
                                  float* __restrict__ agg) {
  int tid = blockIdx.x * blockDim.x + threadIdx.x;
  if (tid >= N_EDGES * RM) return;
  int e = tid / RM, rm = tid % RM;
  int tgt = (int)ei[e];
  int src = (int)ei[N_EDGES + e];
  float cr = conn[e * 2], ci = conn[e * 2 + 1];
  float pr = pscat[(e * RM + rm) * 2], pi = pscat[(e * RM + rm) * 2 + 1];
  const float* xs = x + (size_t)src * C_IN * 2;
  float* ag = agg + ((size_t)tgt * KCONV + rm * C_IN) * 2;
#pragma unroll 4
  for (int c = 0; c < C_IN; ++c) {
    float xr = xs[c * 2], xi = xs[c * 2 + 1];
    float yr = cr * xr - ci * xi;
    float yi = cr * xi + ci * xr;
    atomicAdd(&ag[c * 2],     pr * yr - pi * yi);
    atomicAdd(&ag[c * 2 + 1], pr * yi + pi * yr);
  }
}

// ---------------- conv contraction (96 -> 16) + modReLU ---------------------
__global__ void conv_contract(const float* __restrict__ agg,
                              const float* __restrict__ Wconv,
                              const float* __restrict__ bnl1,
                              float* __restrict__ x1) {
  int tid = blockIdx.x * blockDim.x + threadIdx.x;
  if (tid >= N_NODES * C_IN) return;
  int n = tid / C_IN, o = tid % C_IN;
  const float* a = agg + (size_t)n * KCONV * 2;
  float sr = 0.f, si = 0.f;
#pragma unroll 8
  for (int k = 0; k < KCONV; ++k) {
    float w = Wconv[k * C_IN + o];
    sr += a[k * 2] * w;
    si += a[k * 2 + 1] * w;
  }
  float mag = sqrtf(sr * sr + si * si + EPSV);
  float s = fmaxf(mag + bnl1[o], 0.f) / mag;
  x1[tid * 2] = sr * s;
  x1[tid * 2 + 1] = si * s;
}

// ---------------- TangentPerceptron: complex linear + modReLU ---------------
__global__ void tron(const float* __restrict__ x1,
                     const float* __restrict__ Wtron,
                     const float* __restrict__ bnl2,
                     float* __restrict__ xt) {
  int tid = blockIdx.x * blockDim.x + threadIdx.x;
  if (tid >= N_NODES * N_DES) return;
  int n = tid / N_DES, d = tid % N_DES;
  const float* xs = x1 + (size_t)n * C_IN * 2;
  float sr = 0.f, si = 0.f;
#pragma unroll
  for (int c = 0; c < C_IN; ++c) {
    float xr = xs[c * 2], xi = xs[c * 2 + 1];
    float wr = Wtron[(c * N_DES + d) * 2], wi = Wtron[(c * N_DES + d) * 2 + 1];
    sr += xr * wr - xi * wi;
    si += xr * wi + xi * wr;
  }
  float mag = sqrtf(sr * sr + si * si + EPSV);
  float s = fmaxf(mag + bnl2[d], 0.f) / mag;
  xt[tid * 2] = sr * s;
  xt[tid * 2 + 1] = si * s;
}

// ---------------- ECHO edge scatter (dominant atomic phase) -----------------
// thread = (edge, d). hist[tgt,d,b,:] += pcmp_echo[e,b] * (conn[e]*xt[src,d])
__global__ void echo_scatter(const float* __restrict__ xt,
                             const long long* __restrict__ ei,
                             const float* __restrict__ pecho,
                             const float* __restrict__ conn,
                             float* __restrict__ hist) {
  int tid = blockIdx.x * blockDim.x + threadIdx.x;
  if (tid >= N_EDGES * N_DES) return;
  int e = tid / N_DES, d = tid % N_DES;
  int tgt = (int)ei[e];
  int src = (int)ei[N_EDGES + e];
  float cr = conn[e * 2], ci = conn[e * 2 + 1];
  float xr = xt[((size_t)src * N_DES + d) * 2];
  float xi = xt[((size_t)src * N_DES + d) * 2 + 1];
  float yr = cr * xr - ci * xi;
  float yi = cr * xi + ci * xr;
  float* hp = hist + ((size_t)tgt * N_DES + d) * NBINS * 2;
  const float* pe = pecho + (size_t)e * NBINS;
#pragma unroll 5
  for (int b = 0; b < NBINS; ++b) {
    float w = pe[b];
    atomicAdd(&hp[b * 2],     w * yr);
    atomicAdd(&hp[b * 2 + 1], w * yi);
  }
}

// ---------------- descriptor: |hist| -> f16, K-padded -----------------------
__global__ void make_desc(const float* __restrict__ hist, _Float16* __restrict__ descf) {
  int tid = blockIdx.x * blockDim.x + threadIdx.x;
  if (tid >= N_NODES * DBP) return;
  int n = tid / DBP, j = tid % DBP;
  float v = 0.f;
  if (j < DB) {
    float hr = hist[((size_t)n * DB + j) * 2];
    float hi = hist[((size_t)n * DB + j) * 2 + 1];
    v = sqrtf(hr * hr + hi * hi + EPSV);
  }
  descf[(size_t)n * DBP + j] = (_Float16)v;
}

// ---------------- residual: cnorm(clin(x, W_res)) + bL3 ---------------------
__global__ void residual(const float* __restrict__ x, const float* __restrict__ Wres,
                         const float* __restrict__ bL3, float* __restrict__ res) {
  int tid = blockIdx.x * blockDim.x + threadIdx.x;
  if (tid >= N_NODES * OUTD) return;
  int n = tid / OUTD, o = tid % OUTD;
  const float* xs = x + (size_t)n * C_IN * 2;
  float sr = 0.f, si = 0.f;
#pragma unroll
  for (int c = 0; c < C_IN; ++c) {
    float xr = xs[c * 2], xi = xs[c * 2 + 1];
    float wr = Wres[(c * OUTD + o) * 2], wi = Wres[(c * OUTD + o) * 2 + 1];
    sr += xr * wr - xi * wi;
    si += xr * wi + xi * wr;
  }
  res[tid] = sqrtf(sr * sr + si * si + EPSV) + bL3[o];
}

// ---------------- WMMA fragment loaders (contiguous k order, same in A and B) ----------------
// A (16xK f16 row-major, stride in halfs): lane m=l&15 kg=l>>4,
// halves 0-15 = K[kk+kg*16 ..] (contiguous, matches load_b16)
__device__ __forceinline__ v16h load_a16(const _Float16* base, int stride, int kk, int lane) {
  int m = lane & 15, kg = lane >> 4;
  const _Float16* p = base + (size_t)m * stride + kk + kg * 16;
  v8h lo = *(const v8h*)p;
  v8h hi = *(const v8h*)(p + 8);
  v16h a;
#pragma unroll
  for (int i = 0; i < 8; ++i) { a[i] = lo[i]; a[i + 8] = hi[i]; }
  return a;
}
// B (Kx16): weights pre-transposed so lane's column is a contiguous K-run.
// lane col = l&15, K = kg*16 + h  -> one 32B load.
__device__ __forceinline__ v16h load_b16(const _Float16* wt, int stride, int colbase,
                                         int kk, int lane) {
  int n = lane & 15, kg = lane >> 4;
  return *(const v16h*)(wt + (size_t)(colbase + n) * stride + kk + kg * 16);
}

// ---------------- MLP head: 3 GEMMs via v_wmma_f32_16x16x32_f16 -------------
__global__ __launch_bounds__(512) void mlp_head(const _Float16* __restrict__ descf,
                                                const _Float16* __restrict__ W1t,
                                                const _Float16* __restrict__ W2t,
                                                const _Float16* __restrict__ W3t,
                                                const float* __restrict__ bL1,
                                                const float* __restrict__ bL2,
                                                const float* __restrict__ res,
                                                float* __restrict__ out) {
  __shared__ __align__(32) _Float16 h1s[16 * H1D];
  __shared__ __align__(32) _Float16 h2s[16 * H2D];
  const int lane = threadIdx.x & 31;
  const int w = threadIdx.x >> 5;          // wave id 0..15
  const int tile = blockIdx.x;             // 625 tiles of 16 nodes
  const _Float16* atile = descf + (size_t)tile * 16 * DBP;

  // stage 1: h1[16x256] = relu(desc[16x416] @ W1 + bL1); wave w -> cols [16w,16w+16)
  {
    v8f c = {};
    for (int kk = 0; kk < DBP; kk += 32) {
      v16h a = load_a16(atile, DBP, kk, lane);
      v16h b = load_b16(W1t, DBP, w * 16, kk, lane);
      c = __builtin_amdgcn_wmma_f32_16x16x32_f16(false, a, false, b, (short)0, c, false, false);
    }
    int n = w * 16 + (lane & 15);
    int mb = (lane >> 4) * 8;
    float bias = bL1[n];
#pragma unroll
    for (int r = 0; r < 8; ++r)
      h1s[(mb + r) * H1D + n] = (_Float16)fmaxf(c[r] + bias, 0.f);
  }
  __syncthreads();

  // stage 2: h2[16x128] = relu(h1 @ W2 + bL2); waves 0..7 (wave-uniform branch)
  if (w < 8) {
    v8f c = {};
    for (int kk = 0; kk < H1D; kk += 32) {
      v16h a = load_a16(h1s, H1D, kk, lane);
      v16h b = load_b16(W2t, H1D, w * 16, kk, lane);
      c = __builtin_amdgcn_wmma_f32_16x16x32_f16(false, a, false, b, (short)0, c, false, false);
    }
    int n = w * 16 + (lane & 15);
    int mb = (lane >> 4) * 8;
    float bias = bL2[n];
#pragma unroll
    for (int r = 0; r < 8; ++r)
      h2s[(mb + r) * H2D + n] = (_Float16)fmaxf(c[r] + bias, 0.f);
  }
  __syncthreads();

  // stage 3: out[16x32] = h2 @ W3 + res (res already holds bL3 + residual norm)
  if (w < 2) {
    v8f c = {};
    for (int kk = 0; kk < H2D; kk += 32) {
      v16h a = load_a16(h2s, H2D, kk, lane);
      v16h b = load_b16(W3t, H2D, w * 16, kk, lane);
      c = __builtin_amdgcn_wmma_f32_16x16x32_f16(false, a, false, b, (short)0, c, false, false);
    }
    int n = w * 16 + (lane & 15);
    int mb = (lane >> 4) * 8;
#pragma unroll
    for (int r = 0; r < 8; ++r) {
      size_t row = (size_t)tile * 16 + mb + r;
      out[row * OUTD + n] = c[r] + res[row * OUTD + n];
    }
  }
}

// ---------------- launch ----------------------------------------------------
extern "C" void kernel_launch(void* const* d_in, const int* in_sizes, int n_in,
                              void* d_out, int out_size, void* d_ws, size_t ws_size,
                              hipStream_t stream) {
  const float*     x     = (const float*)d_in[0];
  const long long* ei    = (const long long*)d_in[1];
  const float*     pscat = (const float*)d_in[2];
  const float*     pecho = (const float*)d_in[3];
  const float*     conn  = (const float*)d_in[4];
  const float*     Wconv = (const float*)d_in[5];
  const float*     bnl1  = (const float*)d_in[6];
  const float*     Wtron = (const float*)d_in[7];
  const float*     bnl2  = (const float*)d_in[8];
  const float*     W1    = (const float*)d_in[9];
  const float*     bL1   = (const float*)d_in[10];
  const float*     W2    = (const float*)d_in[11];
  const float*     bL2   = (const float*)d_in[12];
  const float*     W3    = (const float*)d_in[13];
  const float*     bL3   = (const float*)d_in[14];
  const float*     Wres  = (const float*)d_in[15];
  float* out = (float*)d_out;

  char* ws = (char*)d_ws;
  float*    agg   = (float*)(ws + WS_AGG);
  float*    hist  = (float*)(ws + WS_HIST);
  float*    x1    = (float*)(ws + WS_X1);
  float*    xt    = (float*)(ws + WS_XT);
  float*    res   = (float*)(ws + WS_RES);
  _Float16* descf = (_Float16*)(ws + WS_DESC);
  _Float16* W1t   = (_Float16*)(ws + WS_W1T);
  _Float16* W2t   = (_Float16*)(ws + WS_W2T);
  _Float16* W3t   = (_Float16*)(ws + WS_W3T);

  // zero the accumulators every launch (deterministic across graph replays)
  hipMemsetAsync(ws, 0, WS_ZERO_BYTES, stream);

  const int wprep = H1D * DBP + H2D * H1D + OUTD * H2D;   // 143360
  prep_weights<<<(wprep + 255) / 256, 256, 0, stream>>>(W1, W2, W3, W1t, W2t, W3t);

  fieldconv_scatter<<<(N_EDGES * RM + 255) / 256, 256, 0, stream>>>(x, ei, pscat, conn, agg);
  conv_contract<<<(N_NODES * C_IN + 255) / 256, 256, 0, stream>>>(agg, Wconv, bnl1, x1);
  tron<<<(N_NODES * N_DES + 255) / 256, 256, 0, stream>>>(x1, Wtron, bnl2, xt);
  echo_scatter<<<(N_EDGES * N_DES + 255) / 256, 256, 0, stream>>>(xt, ei, pecho, conn, hist);
  make_desc<<<(N_NODES * DBP + 255) / 256, 256, 0, stream>>>(hist, descf);
  residual<<<(N_NODES * OUTD + 255) / 256, 256, 0, stream>>>(x, Wres, bL3, res);

  mlp_head<<<N_NODES / 16, 512, 0, stream>>>(descf, W1t, W2t, W3t, bL1, bL2, res, out);
}
